// ColBERTModel_18262200943121
// MI455X (gfx1250) — matrix operand order repaired and checked by reading, not compiled, not run on hardware
//
#include <hip/hip_runtime.h>

typedef __attribute__((ext_vector_type(16))) _Float16 v16h;
typedef __attribute__((ext_vector_type(8)))  float    v8f;

#define HID   128
#define TTOK  64
#define LDA   136   // padded f16 row stride (272 B: 16B-aligned, bank rotation 4)
#define LDH   132   // padded f32 row stride (528 B: 16B-aligned, bank rotation 4)
#define NEGF  (-1e30f)

// ---------------- monotonic float<->uint keys (order-preserving, handles negatives) ----
__device__ __forceinline__ unsigned fkey(float f) {
    unsigned b = __float_as_uint(f);
    return (b & 0x80000000u) ? ~b : (b | 0x80000000u);
}
__device__ __forceinline__ float funkey(unsigned k) {
    unsigned b = (k & 0x80000000u) ? (k ^ 0x80000000u) : ~k;
    return __uint_as_float(b);
}
__device__ __forceinline__ unsigned pack2h(float a, float b) {
    union { _Float16 h[2]; unsigned u; } x;
    x.h[0] = (_Float16)a; x.h[1] = (_Float16)b;
    return x.u;
}

// ---------------- shared-memory layout for one 64-token encode tile ----------------
struct EncShared {
    alignas(16) _Float16 w1[HID * LDA];   // 34 KB staged f16 W1^T (row = out col, padded)
    alignas(16) _Float16 w2[HID * LDA];   // 34 KB staged f16 W2^T
    alignas(16) _Float16 act[TTOK * LDA]; // 17 KB f16 activations (x -> h1 -> out), padded rows
    alignas(16) float    h32[TTOK * LDH]; // 33 KB f32 pre-LN buffer, padded rows
    float    ps[TTOK * 4];
    float    pq[TTOK * 4];
    float    mean[TTOK];
    float    inv[TTOK];
    int      msk[TTOK];
    unsigned qkey[TTOK];                  // per-query running max keys (doc kernel)
};

// ---------------- WMMA fragment loaders (16x16x32 f16, wave32) ----------------
// A (16x32, MxK): lane m=l&15, half=l>>4; elements = two contiguous 8-half chunks:
//   e=0..7  -> K = k0 + 8*hf + e
//   e=8..15 -> K = k0 + 16 + 8*hf + (e-8)
__device__ __forceinline__ v16h load_a_frag(const _Float16* A, int lda, int row0, int k0, int lane) {
    int m = lane & 15, hf = lane >> 4;
    const _Float16* p = A + (row0 + m) * lda + k0 + 8 * hf;
    v16h a;
#pragma unroll
    for (int e = 0; e < 8; ++e)  a[e] = p[e];
#pragma unroll
    for (int e = 0; e < 8; ++e)  a[8 + e] = p[16 + e];
    return a;
}
// B (32x16, KxN) with B[k][n] = M[n][k] (M row-major, e.g. W^T or qv):
//   same K order as A: e=0..7 -> K = k0 + 8*hf + e, e=8..15 -> K = k0 + 16 + 8*hf + (e-8)
__device__ __forceinline__ v16h load_bt_frag(const _Float16* M, int ldm, int k0, int col0, int lane) {
    int n = lane & 15, hf = lane >> 4;
    const _Float16* p = M + (col0 + n) * ldm + k0 + 8 * hf;
    v16h b;
#pragma unroll
    for (int e = 0; e < 8; ++e)  b[e] = p[e];
#pragma unroll
    for (int e = 0; e < 8; ++e)  b[8 + e] = p[16 + e];
    return b;
}

// ---------------- one GEMM: act[64x128] @ W + bias -> relu -> h32 (W staged as W^T) ----
__device__ __forceinline__ void gemm_relu(EncShared& sh, const _Float16* WTh, const float* bias, int tid) {
    int w = tid >> 5, lane = tid & 31;
    v8f acc[4];
#pragma unroll
    for (int mt = 0; mt < 4; ++mt) acc[mt] = (v8f){};
#pragma unroll
    for (int k0 = 0; k0 < HID; k0 += 32) {
        v16h bf = load_bt_frag(WTh, LDA, k0, w * 16, lane);
#pragma unroll
        for (int mt = 0; mt < 4; ++mt) {
            v16h af = load_a_frag(sh.act, LDA, mt * 16, k0, lane);
            acc[mt] = __builtin_amdgcn_wmma_f32_16x16x32_f16(false, af, false, bf,
                                                             (short)0, acc[mt], false, false);
        }
    }
    int n = lane & 15, hf = lane >> 4;
    int col = w * 16 + n;
    float bv = bias[col];
#pragma unroll
    for (int mt = 0; mt < 4; ++mt)
#pragma unroll
        for (int v = 0; v < 8; ++v) {
            int row = mt * 16 + v + 8 * hf;
            sh.h32[row * LDH + col] = fmaxf(acc[mt][v] + bv, 0.0f);
        }
}

// ---------------- LayerNorm over h32 rows -> f16 act (optionally * mask) ----------------
__device__ __forceinline__ void layernorm_to_f16(EncShared& sh, const float* lnw, const float* lnb,
                                                 bool apply_mask, int tid) {
    int r = tid >> 2, s4 = tid & 3;
    const float* hr = &sh.h32[r * LDH + s4 * 32];
    float sm = 0.f, sq = 0.f;
#pragma unroll
    for (int j = 0; j < 32; ++j) { float x = hr[j]; sm += x; sq += x * x; }
    sh.ps[r * 4 + s4] = sm; sh.pq[r * 4 + s4] = sq;
    __syncthreads();
    if (tid < TTOK) {
        float s = sh.ps[tid*4] + sh.ps[tid*4+1] + sh.ps[tid*4+2] + sh.ps[tid*4+3];
        float q = sh.pq[tid*4] + sh.pq[tid*4+1] + sh.pq[tid*4+2] + sh.pq[tid*4+3];
        float mean = s * (1.0f / HID);
        float var  = q * (1.0f / HID) - mean * mean;
        sh.mean[tid] = mean;
        sh.inv[tid]  = rsqrtf(var + 1e-5f);
    }
    __syncthreads();
    float mean = sh.mean[r], inv = sh.inv[r];
    float mf = apply_mask ? (sh.msk[r] ? 1.0f : 0.0f) : 1.0f;
    unsigned* dst = (unsigned*)&sh.act[r * LDA + s4 * 32];
#pragma unroll
    for (int j = 0; j < 16; ++j) {
        int c0 = s4 * 32 + 2 * j;
        float a = (sh.h32[r * LDH + c0]     - mean) * inv * lnw[c0]     + lnb[c0];
        float b = (sh.h32[r * LDH + c0 + 1] - mean) * inv * lnw[c0 + 1] + lnb[c0 + 1];
        dst[j] = pack2h(a * mf, b * mf);
    }
    __syncthreads();
}

// ---------------- full encode of a 64-token tile; result in sh.act (f16) ----------------
__device__ __forceinline__ void encode_tile(const int* ids, const int* mask, const float* emb,
                                            const _Float16* W1t, const float* b1,
                                            const _Float16* W2t, const float* b2,
                                            const float* lnw, const float* lnb,
                                            EncShared& sh, int tid) {
    // stage transposed weights into padded LDS rows (dword copies)
    {
        const unsigned* s1 = (const unsigned*)W1t; unsigned* d1 = (unsigned*)sh.w1;
        const unsigned* s2 = (const unsigned*)W2t; unsigned* d2 = (unsigned*)sh.w2;
#pragma unroll 4
        for (int i = 0; i < 32; ++i) {
            int idx = i * 256 + tid;            // 0..8191 source dwords (64 dwords/row)
            int row = idx >> 6, c = idx & 63;
            int dsti = row * (LDA / 2) + c;     // padded: 68 dwords/row
            d1[dsti] = s1[idx];
            d2[dsti] = s2[idx];
        }
    }
    if (tid < TTOK) sh.msk[tid] = mask[tid];
    // gather embeddings: 4 threads per token, 32 floats each, convert to f16
    {
        int t = tid >> 2, s4 = tid & 3;
        int id = ids[t];
        const float4* src = (const float4*)(emb + (size_t)id * HID + s4 * 32);
        unsigned* dst = (unsigned*)&sh.act[t * LDA + s4 * 32];
#pragma unroll
        for (int j = 0; j < 8; ++j) {
            float4 v = src[j];
            dst[2 * j]     = pack2h(v.x, v.y);
            dst[2 * j + 1] = pack2h(v.z, v.w);
        }
    }
    __syncthreads();
    gemm_relu(sh, sh.w1, b1, tid);
    __syncthreads();
    layernorm_to_f16(sh, lnw, lnb, /*mask*/false, tid);
    gemm_relu(sh, sh.w2, b2, tid);
    __syncthreads();
    layernorm_to_f16(sh, lnw, lnb, /*mask*/true, tid);
}

// ---------------- kernels ----------------
// Writes W^T in f16: wdst[m][n*128 + k] = W_m[k][n]
__global__ void convert_weights(const float* __restrict__ qW1, const float* __restrict__ qW2,
                                const float* __restrict__ dW1, const float* __restrict__ dW2,
                                _Float16* __restrict__ wdst) {
    int i = blockIdx.x * blockDim.x + threadIdx.x;     // 0..65535
    int m = i >> 14, r = i & 16383;                    // r = n*128 + k
    const float* src = (m == 0) ? qW1 : (m == 1) ? qW2 : (m == 2) ? dW1 : dW2;
    wdst[i] = (_Float16)src[((r & 127) << 7) + (r >> 7)];  // W[k][n]
}

__global__ void init_gkey(unsigned* __restrict__ g) {
    int i = blockIdx.x * blockDim.x + threadIdx.x;
    if (i < 64 * 64) g[i] = fkey(NEGF);
}

__global__ __launch_bounds__(256) void query_encode(
    const int* __restrict__ qids, const int* __restrict__ qmask, const float* __restrict__ qemb,
    const _Float16* __restrict__ w4h, const float* __restrict__ qb1, const float* __restrict__ qb2,
    const float* __restrict__ qlnw, const float* __restrict__ qlnb, _Float16* __restrict__ qvh) {
    __shared__ EncShared sh;
    int b = blockIdx.x, tid = threadIdx.x;
    encode_tile(qids + b * 64, qmask + b * 64, qemb,
                w4h, qb1, w4h + 16384, qb2, qlnw, qlnb, sh, tid);
    // copy padded LDS act -> dense global qv[b] (dword granularity)
    const unsigned* s = (const unsigned*)sh.act;
    unsigned* d = (unsigned*)(qvh + (size_t)b * TTOK * HID);
#pragma unroll
    for (int i = 0; i < 16; ++i) {
        int idx = i * 256 + tid;            // 0..4095 dense dwords (64/row)
        int row = idx >> 6, c = idx & 63;
        d[idx] = s[row * (LDA / 2) + c];
    }
}

__global__ __launch_bounds__(256) void doc_sim(
    const int* __restrict__ dids, const int* __restrict__ dmask, const float* __restrict__ demb,
    const _Float16* __restrict__ w4h, const float* __restrict__ db1, const float* __restrict__ db2,
    const float* __restrict__ dlnw, const float* __restrict__ dlnb,
    const _Float16* __restrict__ qvh, unsigned* __restrict__ gkey) {
    __shared__ EncShared sh;
    int tile = blockIdx.x, b = blockIdx.y, tid = threadIdx.x;
    if (tid < TTOK) sh.qkey[tid] = fkey(NEGF);
    encode_tile(dids + b * 4096 + tile * 64, dmask + b * 4096 + tile * 64, demb,
                w4h + 2 * 16384, db1, w4h + 3 * 16384, db2, dlnw, dlnb, sh, tid);
    // sim tile: D = dv[64x128] @ qv^T; rows (M) = docs, cols (N) = queries
    int w = tid >> 5, lane = tid & 31, n = lane & 15, hf = lane >> 4;
    const _Float16* qb = qvh + (size_t)b * TTOK * HID;
#pragma unroll
    for (int i = 0; i < 2; ++i) {
        int t = 2 * w + i;
        int mt = t >> 2;   // doc 16-row tile
        int qt = t & 3;    // query 16-col tile
        v8f c = (v8f){};
#pragma unroll
        for (int k0 = 0; k0 < HID; k0 += 32) {
            v16h af = load_a_frag(sh.act, LDA, mt * 16, k0, lane);
            v16h bf = load_bt_frag(qb, HID, k0, qt * 16, lane);
            c = __builtin_amdgcn_wmma_f32_16x16x32_f16(false, af, false, bf,
                                                       (short)0, c, false, false);
        }
        float lmax = NEGF;
#pragma unroll
        for (int v = 0; v < 8; ++v) {
            int d = mt * 16 + v + 8 * hf;
            float s = sh.msk[d] ? c[v] : NEGF;
            lmax = fmaxf(lmax, s);
        }
        atomicMax(&sh.qkey[qt * 16 + n], fkey(lmax));
    }
    __syncthreads();
    if (tid < TTOK) atomicMax(&gkey[b * 64 + tid], sh.qkey[tid]);
}

__global__ __launch_bounds__(64) void reduce_sum(const unsigned* __restrict__ gkey,
                                                 float* __restrict__ out) {
    __shared__ float s[64];
    int b = blockIdx.x, q = threadIdx.x;
    s[q] = funkey(gkey[b * 64 + q]);
    __syncthreads();
    if (q == 0) {
        float t = 0.f;
#pragma unroll
        for (int i = 0; i < 64; ++i) t += s[i];
        out[b] = t;
    }
}

// ---------------- launch ----------------
extern "C" void kernel_launch(void* const* d_in, const int* in_sizes, int n_in,
                              void* d_out, int out_size, void* d_ws, size_t ws_size,
                              hipStream_t stream) {
    const int*   qids  = (const int*)d_in[0];
    const int*   dids  = (const int*)d_in[1];
    const int*   qmask = (const int*)d_in[2];
    const int*   dmask = (const int*)d_in[3];
    const float* qemb  = (const float*)d_in[4];
    const float* qW1   = (const float*)d_in[5];
    const float* qb1   = (const float*)d_in[6];
    const float* qW2   = (const float*)d_in[7];
    const float* qb2   = (const float*)d_in[8];
    const float* qlnw  = (const float*)d_in[9];
    const float* qlnb  = (const float*)d_in[10];
    const float* demb  = (const float*)d_in[11];
    const float* dW1   = (const float*)d_in[12];
    const float* db1   = (const float*)d_in[13];
    const float* dW2   = (const float*)d_in[14];
    const float* db2   = (const float*)d_in[15];
    const float* dlnw  = (const float*)d_in[16];
    const float* dlnb  = (const float*)d_in[17];
    float* out = (float*)d_out;

    char* ws = (char*)d_ws;
    _Float16* w4h  = (_Float16*)ws;                              // 4*128*128 f16 = 131072 B (transposed)
    _Float16* qvh  = (_Float16*)(ws + 131072);                   // 64*64*128 f16 = 1048576 B
    unsigned* gkey = (unsigned*)(ws + 131072 + 1048576);         // 64*64 u32 = 16384 B

    convert_weights<<<256, 256, 0, stream>>>(qW1, qW2, dW1, dW2, w4h);
    init_gkey<<<16, 256, 0, stream>>>(gkey);
    query_encode<<<64, 256, 0, stream>>>(qids, qmask, qemb, w4h, qb1, qb2, qlnw, qlnb, qvh);
    doc_sim<<<dim3(64, 64), 256, 0, stream>>>(dids, dmask, demb, w4h, db1, db2, dlnw, dlnb, qvh, gkey);
    reduce_sum<<<64, 64, 0, stream>>>(gkey, out);
}
